// Decoder_48017734369906
// MI455X (gfx1250) — compile-verified, matrix operand order checked by reading, not run on hardware
//
#include <hip/hip_runtime.h>

// ---------------- problem constants ----------------
#define Tt   32
#define Bb   64
#define Ss   100
#define EMB  512
#define ENC  1024
#define DEC  1024
#define ATT  1024
#define G3   3072          // 3*DEC
#define L0   1536          // EMB+ENC
#define RO   2560          // EMB+DEC+ENC (readout cat)
#define CP   2048          // DEC+ENC (copy gate cat)

// output offsets (floats) in d_out, concat in return order
#define OFF_G   0u                                   // g_outputs  [T,B,512]
#define OFF_C   (Tt*Bb*512u)                         // c_outputs  [T,B,S]
#define OFF_CP  (OFF_C + Tt*Bb*(unsigned)Ss)         // copy_gates [T,B,1]
#define OFF_H   (OFF_CP + Tt*Bb)                     // hid_f      [2,B,DEC]
#define OFF_CL  (OFF_H + 2u*Bb*DEC)                  // c_outputs[-1] [B,S]
#define OFF_CF  (OFF_CL + Bb*(unsigned)Ss)           // ctx_f      [B,ENC]

typedef __attribute__((ext_vector_type(16))) _Float16 v16h;
typedef __attribute__((ext_vector_type(8)))  _Float16 v8h;
typedef __attribute__((ext_vector_type(8)))  float    v8f;

union Frag16 { v16h v; v8h h[2]; };
union FragAcc { v8f v; float f[8]; };

// ---------------- WMMA GEMM: C[M,N] = A[M,K] * W[N,K]^T (+bias) ----------------
// A,W f16 row-major. One wave -> 16x64 tile (4 accumulators). Block=128 -> 16x256.
// Requires M%16==0, N%256==0, K%32==0 (all shapes here satisfy this).
__device__ __forceinline__ void store_tile(v8f acc, const float* bias, float* Cf,
                                           _Float16* Ch, int N, int mo, int cb) {
  FragAcc fa; fa.v = acc;
  float bv = bias ? bias[cb] : 0.0f;
#pragma unroll
  for (int r = 0; r < 8; ++r) {
    float v = fa.f[r] + bv;
    size_t idx = (size_t)(mo + r) * N + cb;
    if (Cf) Cf[idx] = v;
    if (Ch) Ch[idx] = (_Float16)v;
  }
}

__launch_bounds__(128)
__global__ void gemm_wmma(const _Float16* __restrict__ A,
                          const _Float16* __restrict__ W,
                          const float* __restrict__ bias,
                          float* __restrict__ Cf,
                          _Float16* __restrict__ Ch,
                          int M, int N, int K) {
  const int lane = threadIdx.x & 31;
  const int wave = threadIdx.x >> 5;
  const int m0   = blockIdx.y << 4;
  const int n0   = (blockIdx.x << 8) + (wave << 6);
  const int r15  = lane & 15;
  const int kh8  = (lane >> 4) << 3;   // 0 or 8 (K-half per ISA layout)

  const _Float16* ap  = A + (size_t)(m0 + r15) * K + kh8;
  const _Float16* wp0 = W + (size_t)(n0 +  0 + r15) * K + kh8;
  const _Float16* wp1 = W + (size_t)(n0 + 16 + r15) * K + kh8;
  const _Float16* wp2 = W + (size_t)(n0 + 32 + r15) * K + kh8;
  const _Float16* wp3 = W + (size_t)(n0 + 48 + r15) * K + kh8;

  v8f acc0 = {}, acc1 = {}, acc2 = {}, acc3 = {};
  for (int k = 0; k < K; k += 32) {
    Frag16 a, b0, b1, b2, b3;
    a.h[0]  = *(const v8h*)(ap  + k);  a.h[1]  = *(const v8h*)(ap  + k + 16);
    b0.h[0] = *(const v8h*)(wp0 + k);  b0.h[1] = *(const v8h*)(wp0 + k + 16);
    b1.h[0] = *(const v8h*)(wp1 + k);  b1.h[1] = *(const v8h*)(wp1 + k + 16);
    b2.h[0] = *(const v8h*)(wp2 + k);  b2.h[1] = *(const v8h*)(wp2 + k + 16);
    b3.h[0] = *(const v8h*)(wp3 + k);  b3.h[1] = *(const v8h*)(wp3 + k + 16);
    acc0 = __builtin_amdgcn_wmma_f32_16x16x32_f16(false, a.v, false, b0.v, (short)0, acc0, false, false);
    acc1 = __builtin_amdgcn_wmma_f32_16x16x32_f16(false, a.v, false, b1.v, (short)0, acc1, false, false);
    acc2 = __builtin_amdgcn_wmma_f32_16x16x32_f16(false, a.v, false, b2.v, (short)0, acc2, false, false);
    acc3 = __builtin_amdgcn_wmma_f32_16x16x32_f16(false, a.v, false, b3.v, (short)0, acc3, false, false);
  }
  const int mo = m0 + ((lane >> 4) << 3);
  store_tile(acc0, bias, Cf, Ch, N, mo, n0 +  0 + r15);
  store_tile(acc1, bias, Cf, Ch, N, mo, n0 + 16 + r15);
  store_tile(acc2, bias, Cf, Ch, N, mo, n0 + 32 + r15);
  store_tile(acc3, bias, Cf, Ch, N, mo, n0 + 48 + r15);
}

// ---------------- elementwise / helper kernels ----------------
__global__ void k_f32_to_f16(const float* __restrict__ s, _Float16* __restrict__ d, int n) {
  int i = blockIdx.x * blockDim.x + threadIdx.x;
  if (i < n) d[i] = (_Float16)s[i];
}
__global__ void k_f32_copy(const float* __restrict__ s, float* __restrict__ d, int n) {
  int i = blockIdx.x * blockDim.x + threadIdx.x;
  if (i < n) d[i] = s[i];
}
// context [S,B,ENC] f32 -> ctx_bt [B,S,ENC] f16
__global__ void k_ctx_transpose(const float* __restrict__ ctx, _Float16* __restrict__ d) {
  int i = blockIdx.x * blockDim.x + threadIdx.x;
  if (i >= Bb * Ss * ENC) return;
  int e = i % ENC, r = i / ENC;
  int s = r % Ss,  b = r / Ss;
  d[i] = (_Float16)ctx[(size_t)s * Bb * ENC + (size_t)b * ENC + e];
}
// x = [emb_t | att] (f16), [B, L0]
__global__ void k_pack_x(const _Float16* __restrict__ embt, const _Float16* __restrict__ atth,
                         _Float16* __restrict__ x) {
  int i = blockIdx.x * blockDim.x + threadIdx.x;
  if (i >= Bb * L0) return;
  int c = i % L0, b = i / L0;
  x[i] = (c < EMB) ? embt[b * EMB + c] : atth[b * ENC + (c - EMB)];
}
// ro = [emb_t | h1 | att] (f16), [B, RO]
__global__ void k_pack_ro(const _Float16* __restrict__ embt, const _Float16* __restrict__ h1h,
                          const _Float16* __restrict__ atth, _Float16* __restrict__ ro) {
  int i = blockIdx.x * blockDim.x + threadIdx.x;
  if (i >= Bb * RO) return;
  int c = i % RO, b = i / RO;
  _Float16 v;
  if (c < EMB)            v = embt[b * EMB + c];
  else if (c < EMB + DEC) v = h1h[b * DEC + (c - EMB)];
  else                    v = atth[b * ENC + (c - EMB - DEC)];
  ro[i] = v;
}
// GRU combine: h = (1-z)*n + z*h_prev ; updates h (f32) and hh (f16) in place
__global__ void k_gru(const float* __restrict__ gi, const float* __restrict__ gh,
                      float* __restrict__ h, _Float16* __restrict__ hh) {
  int i = blockIdx.x * blockDim.x + threadIdx.x;
  if (i >= Bb * DEC) return;
  int j = i % DEC, b = i / DEC;
  size_t base = (size_t)b * G3;
  float ir = gi[base + j], iz = gi[base + DEC + j], in = gi[base + 2 * DEC + j];
  float hr = gh[base + j], hz = gh[base + DEC + j], hn = gh[base + 2 * DEC + j];
  float r = 1.0f / (1.0f + __expf(-(ir + hr)));
  float z = 1.0f / (1.0f + __expf(-(iz + hz)));
  float n = tanhf(in + r * hn);
  float hv = (1.0f - z) * n + z * h[i];
  h[i] = hv;
  hh[i] = (_Float16)hv;
}
// energy[b,s] = sum_a Wv[a]*tanh(pre[b,s,a] + tgt[b,a]) ; masked. One wave per (b,s).
__launch_bounds__(256)
__global__ void k_energy(const _Float16* __restrict__ pre, const float* __restrict__ tgt,
                         const float* __restrict__ Wv, const float* __restrict__ mask,
                         float* __restrict__ energy) {
  int bs = blockIdx.x * 8 + (threadIdx.x >> 5);
  int lane = threadIdx.x & 31;
  int b = bs / Ss;
  const _Float16* p = pre + (size_t)bs * ATT;
  const float* q = tgt + (size_t)b * ATT;
  float sum = 0.0f;
  for (int a = lane; a < ATT; a += 32)
    sum += Wv[a] * tanhf((float)p[a] + q[a]);
#pragma unroll
  for (int off = 16; off; off >>= 1) sum += __shfl_xor(sum, off, 32);
  if (lane == 0) {
    float m = mask[bs];
    energy[bs] = sum * (1.0f - m) + m * (-1.0e6f);
  }
}
// softmax over S per batch row; also writes c_outputs[t]
__launch_bounds__(128)
__global__ void k_softmax(const float* __restrict__ energy, float* __restrict__ score,
                          float* __restrict__ cout) {
  __shared__ float sh[128];
  int b = blockIdx.x, tid = threadIdx.x;
  float v = (tid < Ss) ? energy[b * Ss + tid] : -3.4e38f;
  sh[tid] = v; __syncthreads();
  for (int off = 64; off; off >>= 1) { if (tid < off) sh[tid] = fmaxf(sh[tid], sh[tid + off]); __syncthreads(); }
  float mx = sh[0]; __syncthreads();
  float e = (tid < Ss) ? __expf(v - mx) : 0.0f;
  sh[tid] = e; __syncthreads();
  for (int off = 64; off; off >>= 1) { if (tid < off) sh[tid] += sh[tid + off]; __syncthreads(); }
  float inv = 1.0f / sh[0];
  if (tid < Ss) {
    float sc = e * inv;
    score[b * Ss + tid] = sc;
    cout[b * Ss + tid]  = sc;
  }
}
// new_ctx[b,e] = sum_s score[b,s] * context[s,b,e]
__global__ void k_ctx(const float* __restrict__ score, const float* __restrict__ context,
                      float* __restrict__ attf, _Float16* __restrict__ atth) {
  int e = blockIdx.x * blockDim.x + threadIdx.x;
  int b = blockIdx.y;
  if (e >= ENC) return;
  float acc = 0.0f;
  for (int s = 0; s < Ss; ++s)
    acc += score[b * Ss + s] * context[(size_t)s * Bb * ENC + (size_t)b * ENC + e];
  attf[b * ENC + e] = acc;
  atth[b * ENC + e] = (_Float16)acc;
}
// copy gate: sigmoid([h1|ctx] @ Wcp^T + bcp), one wave per b
__launch_bounds__(64)
__global__ void k_copygate(const float* __restrict__ h1f, const float* __restrict__ attf,
                           const float* __restrict__ Wcp, const float* __restrict__ bcp,
                           float* __restrict__ out) {
  int b = blockIdx.x * 2 + (threadIdx.x >> 5);
  int lane = threadIdx.x & 31;
  float sum = 0.0f;
  for (int j = lane; j < CP; j += 32) {
    float x = (j < DEC) ? h1f[b * DEC + j] : attf[b * ENC + (j - DEC)];
    sum += x * Wcp[j];
  }
#pragma unroll
  for (int off = 16; off; off >>= 1) sum += __shfl_xor(sum, off, 32);
  if (lane == 0) out[b] = 1.0f / (1.0f + __expf(-(sum + bcp[0])));
}
// maxout pairs: g[b,j] = max(readout[b,2j], readout[b,2j+1])
__global__ void k_maxout(const float* __restrict__ ro, float* __restrict__ g) {
  int i = blockIdx.x * blockDim.x + threadIdx.x;
  if (i >= Bb * 512) return;
  int j = i % 512, b = i / 512;
  g[i] = fmaxf(ro[b * DEC + 2 * j], ro[b * DEC + 2 * j + 1]);
}

// ---------------- host ----------------
extern "C" void kernel_launch(void* const* d_in, const int* in_sizes, int n_in,
                              void* d_out, int out_size, void* d_ws, size_t ws_size,
                              hipStream_t stream) {
  (void)in_sizes; (void)n_in; (void)out_size; (void)ws_size;
  const float* output_emb = (const float*)d_in[0];
  const float* hidden     = (const float*)d_in[1];
  const float* context    = (const float*)d_in[2];
  const float* mask       = (const float*)d_in[3];
  const float* init_att   = (const float*)d_in[4];
  const float* w_ih0 = (const float*)d_in[5];
  const float* b_ih0 = (const float*)d_in[6];
  const float* w_hh0 = (const float*)d_in[7];
  const float* b_hh0 = (const float*)d_in[8];
  const float* w_ih1 = (const float*)d_in[9];
  const float* b_ih1 = (const float*)d_in[10];
  const float* w_hh1 = (const float*)d_in[11];
  const float* b_hh1 = (const float*)d_in[12];
  const float* Wc  = (const float*)d_in[13];
  const float* bc  = (const float*)d_in[14];
  const float* Wq  = (const float*)d_in[15];
  const float* Wv  = (const float*)d_in[16];
  const float* Wr  = (const float*)d_in[17];
  const float* br  = (const float*)d_in[18];
  const float* Wcp = (const float*)d_in[19];
  const float* bcp = (const float*)d_in[20];
  float* out = (float*)d_out;

  // workspace bump allocator (256B aligned)
  char* ws = (char*)d_ws;
  size_t off = 0;
  auto alloc = [&](size_t bytes) -> void* {
    void* p = ws + off;
    off = (off + bytes + 255) & ~(size_t)255;
    return p;
  };
  _Float16* w_ih0h = (_Float16*)alloc((size_t)G3 * L0  * 2);
  _Float16* w_hh0h = (_Float16*)alloc((size_t)G3 * DEC * 2);
  _Float16* w_ih1h = (_Float16*)alloc((size_t)G3 * DEC * 2);
  _Float16* w_hh1h = (_Float16*)alloc((size_t)G3 * DEC * 2);
  _Float16* Wqh    = (_Float16*)alloc((size_t)ATT * DEC * 2);
  _Float16* Wch    = (_Float16*)alloc((size_t)ATT * ENC * 2);
  _Float16* Wrh    = (_Float16*)alloc((size_t)DEC * RO  * 2);
  _Float16* emb_h  = (_Float16*)alloc((size_t)Tt * Bb * EMB * 2);
  _Float16* ctxbt  = (_Float16*)alloc((size_t)Bb * Ss * ENC * 2);
  _Float16* pre_h  = (_Float16*)alloc((size_t)Bb * Ss * ATT * 2);
  _Float16* x_h    = (_Float16*)alloc((size_t)Bb * L0 * 2);
  _Float16* ro_h   = (_Float16*)alloc((size_t)Bb * RO * 2);
  float*    gi     = (float*)alloc((size_t)Bb * G3 * 4);
  float*    gh     = (float*)alloc((size_t)Bb * G3 * 4);
  float*    h0f    = (float*)alloc((size_t)Bb * DEC * 4);
  float*    h1f    = (float*)alloc((size_t)Bb * DEC * 4);
  _Float16* h0h    = (_Float16*)alloc((size_t)Bb * DEC * 2);
  _Float16* h1h    = (_Float16*)alloc((size_t)Bb * DEC * 2);
  float*    attf   = (float*)alloc((size_t)Bb * ENC * 4);
  _Float16* atth   = (_Float16*)alloc((size_t)Bb * ENC * 2);
  float*    tgt    = (float*)alloc((size_t)Bb * ATT * 4);
  float*    energy = (float*)alloc((size_t)Bb * Ss * 4);
  float*    score  = (float*)alloc((size_t)Bb * Ss * 4);
  float*    rdout  = (float*)alloc((size_t)Bb * DEC * 4);

  auto conv = [&](const float* s, _Float16* d, int n) {
    k_f32_to_f16<<<(n + 255) / 256, 256, 0, stream>>>(s, d, n);
  };
  auto copy = [&](const float* s, float* d, int n) {
    k_f32_copy<<<(n + 255) / 256, 256, 0, stream>>>(s, d, n);
  };
  auto gemm = [&](const _Float16* A, const _Float16* W_, const float* bias_,
                  float* Cf, _Float16* Ch, int M_, int N_, int K_) {
    dim3 g(N_ / 256, M_ / 16);
    gemm_wmma<<<g, 128, 0, stream>>>(A, W_, bias_, Cf, Ch, M_, N_, K_);
  };

  // ---- one-time conversions (redone each call: deterministic) ----
  conv(w_ih0, w_ih0h, G3 * L0);
  conv(w_hh0, w_hh0h, G3 * DEC);
  conv(w_ih1, w_ih1h, G3 * DEC);
  conv(w_hh1, w_hh1h, G3 * DEC);
  conv(Wq, Wqh, ATT * DEC);
  conv(Wc, Wch, ATT * ENC);
  conv(Wr, Wrh, DEC * RO);
  conv(output_emb, emb_h, Tt * Bb * EMB);
  copy(hidden, h0f, Bb * DEC);
  copy(hidden + (size_t)Bb * DEC, h1f, Bb * DEC);
  conv(hidden, h0h, Bb * DEC);
  conv(hidden + (size_t)Bb * DEC, h1h, Bb * DEC);
  copy(init_att, attf, Bb * ENC);
  conv(init_att, atth, Bb * ENC);
  k_ctx_transpose<<<(Bb * Ss * ENC + 255) / 256, 256, 0, stream>>>(context, ctxbt);

  // precompute[b,s,:] = ctx_bt @ Wc^T + bc   (f16 output)
  gemm(ctxbt, Wch, bc, nullptr, pre_h, Bb * Ss, ATT, ENC);

  // ---- sequential decoder steps ----
  for (int t = 0; t < Tt; ++t) {
    const _Float16* embt = emb_h + (size_t)t * Bb * EMB;
    k_pack_x<<<(Bb * L0 + 255) / 256, 256, 0, stream>>>(embt, atth, x_h);
    gemm(x_h, w_ih0h, b_ih0, gi, nullptr, Bb, G3, L0);
    gemm(h0h, w_hh0h, b_hh0, gh, nullptr, Bb, G3, DEC);
    k_gru<<<(Bb * DEC + 255) / 256, 256, 0, stream>>>(gi, gh, h0f, h0h);
    gemm(h0h, w_ih1h, b_ih1, gi, nullptr, Bb, G3, DEC);
    gemm(h1h, w_hh1h, b_hh1, gh, nullptr, Bb, G3, DEC);
    k_gru<<<(Bb * DEC + 255) / 256, 256, 0, stream>>>(gi, gh, h1f, h1h);
    gemm(h1h, Wqh, nullptr, tgt, nullptr, Bb, ATT, DEC);
    k_energy<<<(Bb * Ss) / 8, 256, 0, stream>>>(pre_h, tgt, Wv, mask, energy);
    k_softmax<<<Bb, 128, 0, stream>>>(energy, score, out + OFF_C + (size_t)t * Bb * Ss);
    k_ctx<<<dim3(ENC / 256, Bb), 256, 0, stream>>>(score, context, attf, atth);
    k_copygate<<<Bb / 2, 64, 0, stream>>>(h1f, attf, Wcp, bcp, out + OFF_CP + (size_t)t * Bb);
    k_pack_ro<<<(Bb * RO + 255) / 256, 256, 0, stream>>>(embt, h1h, atth, ro_h);
    gemm(ro_h, Wrh, br, rdout, nullptr, Bb, DEC, RO);
    k_maxout<<<(Bb * 512 + 255) / 256, 256, 0, stream>>>(rdout, out + OFF_G + (size_t)t * Bb * 512);
  }

  // ---- finalize: hid_f, last attention scores, final ctx ----
  copy(h0f, out + OFF_H, Bb * DEC);
  copy(h1f, out + OFF_H + (size_t)Bb * DEC, Bb * DEC);
  copy(score, out + OFF_CL, Bb * Ss);
  copy(attf, out + OFF_CF, Bb * ENC);
}
